// RPNModel_51513837748298
// MI455X (gfx1250) — matrix operand order repaired and checked by reading, not compiled, not run on hardware
//
#include <hip/hip_runtime.h>
#include <hip/hip_bf16.h>

typedef __attribute__((ext_vector_type(16))) _Float16 v16h;
typedef __attribute__((ext_vector_type(8)))  float    v8f;

#define HFD 48
#define WFD 80
#define CIN 512
#define NF  512
#define BATCHN 16
#define M_TOTAL (BATCHN * HFD * WFD)   /* 61440 */
#define K_CONV  (9 * CIN)              /* 4608  */
#define KANCH 9
#define NPROP (HFD * WFD * KANCH)      /* 34560 */
#define NROI 512
#define NHEAD 64                       /* 18 cls + 36 bbox + 10 pad */
#define HP (HFD + 2)                   /* 50 */
#define WP (WFD + 2)                   /* 82 */

#define LDSA 40                        /* padded LDS row stride in halves (conflict-free) */

/* ---------------- staging / packing kernels ---------------- */

/* features [16][48][80][512] f32 -> zero-padded f16 [16][50][82][512] */
__global__ __launch_bounds__(256) void pad_feat_kernel(const float* __restrict__ src,
                                                       _Float16* __restrict__ dst, int n) {
    int i = blockIdx.x * 256 + threadIdx.x;
    if (i >= n) return;
    int c = i & (CIN - 1);
    int t = i >> 9;
    int xp = t % WP;
    t /= WP;
    int yp = t % HP;
    int b = t / HP;
    float v = 0.f;
    if (yp >= 1 && yp <= HFD && xp >= 1 && xp <= WFD)
        v = src[(((size_t)b * HFD + (yp - 1)) * WFD + (xp - 1)) * CIN + c];
    dst[i] = (_Float16)v;
}

/* conv1_w [3][3][512][512] HWIO -> wT16[f][tap*512+c] (N-major rows, K contiguous) */
__global__ __launch_bounds__(256) void pack_convw_kernel(const float* __restrict__ w,
                                                         _Float16* __restrict__ wT, int n) {
    int i = blockIdx.x * 256 + threadIdx.x;
    if (i < n) {
        int k = i / NF;
        int f = i % NF;
        wT[(size_t)f * K_CONV + k] = (_Float16)w[i];
    }
}

/* cls_w [512][18], bbox_w [512][36] -> hwT16[n][f], n in [0,64), cols 54..63 zero */
__global__ __launch_bounds__(256) void pack_headw_kernel(const float* __restrict__ clsw,
                                                         const float* __restrict__ bboxw,
                                                         _Float16* __restrict__ hwT) {
    int i = blockIdx.x * 256 + threadIdx.x;
    if (i < NF * NHEAD) {
        int f = i / NHEAD, n = i % NHEAD;
        float v = 0.f;
        if (n < 2 * KANCH)      v = clsw[f * (2 * KANCH) + n];
        else if (n < 6 * KANCH) v = bboxw[f * (4 * KANCH) + (n - 2 * KANCH)];
        hwT[(size_t)n * NF + f] = (_Float16)v;
    }
}

/* ---------------- WMMA helpers ---------------- */

__device__ __forceinline__ void store_stage(_Float16* __restrict__ As, _Float16* __restrict__ Bs,
                                            int rA, int offA, int nB, int offB,
                                            uint4 a0, uint4 a1, uint4 b0) {
    *(uint4*)(As + rA * LDSA + offA)     = a0;
    *(uint4*)(As + rA * LDSA + offA + 8) = a1;
    *(uint4*)(Bs + nB * LDSA + offB)     = b0;
}

/* 4 WMMAs: 2 A-frags x 2 B-frags per wave per 32-wide k-step */
__device__ __forceinline__ void compute4(const _Float16* __restrict__ As,
                                         const _Float16* __restrict__ Bs,
                                         int mSubBase, int nSubBase, int laneLo,
                                         int khalfA, int khalfB,
                                         v8f& c00, v8f& c01, v8f& c10, v8f& c11) {
    union { v16h v; uint4 q[2]; } a0, a1, b0, b1;
    const _Float16* ap0 = As + (mSubBase + laneLo) * LDSA + khalfA;
    a0.q[0] = *(const uint4*)ap0;
    a0.q[1] = *(const uint4*)(ap0 + 8);
    const _Float16* ap1 = ap0 + 16 * LDSA;
    a1.q[0] = *(const uint4*)ap1;
    a1.q[1] = *(const uint4*)(ap1 + 8);
    const _Float16* bp0 = Bs + (nSubBase + laneLo) * LDSA + khalfB;
    b0.q[0] = *(const uint4*)bp0;
    b0.q[1] = *(const uint4*)(bp0 + 8);
    const _Float16* bp1 = bp0 + 16 * LDSA;
    b1.q[0] = *(const uint4*)bp1;
    b1.q[1] = *(const uint4*)(bp1 + 8);
    c00 = __builtin_amdgcn_wmma_f32_16x16x32_f16(false, a0.v, false, b0.v, (short)0, c00, false, false);
    c01 = __builtin_amdgcn_wmma_f32_16x16x32_f16(false, a0.v, false, b1.v, (short)0, c01, false, false);
    c10 = __builtin_amdgcn_wmma_f32_16x16x32_f16(false, a1.v, false, b0.v, (short)0, c10, false, false);
    c11 = __builtin_amdgcn_wmma_f32_16x16x32_f16(false, a1.v, false, b1.v, (short)0, c11, false, false);
}

/* ---------------- implicit-GEMM 3x3 conv via WMMA ---------------- */
/* M=61440, K=4608, N=512. 128x64 tile, 8 waves, 4 accs/wave, double-buffered LDS. */

#define NK_CONV (K_CONV / 32)   /* 144 */

__device__ __forceinline__ void load_conv_stage(const _Float16* __restrict__ aBase,
                                                const _Float16* __restrict__ bBase,
                                                int ks, uint4& a0, uint4& a1, uint4& b0) {
    int tap = ks >> 4;                      /* 16 c-steps per tap */
    int qt  = tap / 3;
    int rt  = tap - qt * 3;
    int delta = ((qt - 1) * WP + (rt - 1)) * CIN + ((ks & 15) << 5);
    const _Float16* ap = aBase + delta;     /* zero-padded: no bounds check */
    a0 = *(const uint4*)ap;
    a1 = *(const uint4*)(ap + 8);
    b0 = *(const uint4*)(bBase + (ks << 5));
}

__global__ __launch_bounds__(256) void conv_gemm_kernel(const _Float16* __restrict__ featP,
                                                        const _Float16* __restrict__ wT,
                                                        const float* __restrict__ bias,
                                                        _Float16* __restrict__ h16) {
    __shared__ _Float16 As[2][128 * LDSA];
    __shared__ _Float16 Bs[2][64 * LDSA];

    const int tid   = threadIdx.x;
    const int mBase = blockIdx.x * 128;
    const int nBase = blockIdx.y * 64;

    /* A loader: 2 threads per row, 16 halves each (two b128) */
    const int rA   = tid >> 1;
    const int offA = (tid & 1) * 16;
    const int mRow = mBase + rA;
    const int bb   = mRow / (HFD * WFD);
    const int remA = mRow % (HFD * WFD);
    const int yA   = remA / WFD;
    const int xA   = remA % WFD;
    const _Float16* aBase = featP + (((size_t)bb * HP + yA + 1) * WP + (xA + 1)) * CIN + offA;

    /* B loader: 4 threads per row, 8 halves each */
    const int nB   = tid >> 2;
    const int offB = (tid & 3) * 8;
    const _Float16* bBase = wT + (size_t)(nBase + nB) * K_CONV + offB;

    /* compute mapping: wave -> 32x32 sub-block (2x2 wmma tiles) */
    const int wave     = tid >> 5;
    const int lane     = tid & 31;
    const int mSubBase = (wave & 3) * 32;
    const int nSubBase = (wave >> 2) * 32;
    const int laneLo   = lane & 15;
    const int khalfA   = (lane >> 4) * 16;
    const int khalfB   = (lane >> 4) * 16;

    v8f c00 = {}, c01 = {}, c10 = {}, c11 = {};
    uint4 a0, a1, b0;

    load_conv_stage(aBase, bBase, 0, a0, a1, b0);
    store_stage(As[0], Bs[0], rA, offA, nB, offB, a0, a1, b0);
    __syncthreads();

    for (int ks = 0; ks < NK_CONV; ++ks) {
        const int cur = ks & 1;
        const bool more = (ks + 1) < NK_CONV;
        if (more) load_conv_stage(aBase, bBase, ks + 1, a0, a1, b0);
        compute4(As[cur], Bs[cur], mSubBase, nSubBase, laneLo, khalfA, khalfB, c00, c01, c10, c11);
        if (more) store_stage(As[cur ^ 1], Bs[cur ^ 1], rA, offA, nB, offB, a0, a1, b0);
        __syncthreads();
    }

    /* epilogue: bias + relu -> f16 */
    const int mHalf = (lane >> 4) * 8;
#pragma unroll
    for (int mi = 0; mi < 2; ++mi) {
#pragma unroll
        for (int ni = 0; ni < 2; ++ni) {
            const v8f acc = mi == 0 ? (ni == 0 ? c00 : c01) : (ni == 0 ? c10 : c11);
            const int nOut = nBase + nSubBase + ni * 16 + laneLo;
            const float bn = bias[nOut];
            const int m0 = mBase + mSubBase + mi * 16 + mHalf;
#pragma unroll
            for (int v = 0; v < 8; ++v) {
                float val = acc[v] + bn;
                val = val > 0.f ? val : 0.f;
                h16[(size_t)(m0 + v) * NF + nOut] = (_Float16)val;
            }
        }
    }
}

/* ---------------- fused head GEMM (N=64) + objectness/anchor decode ---------------- */

#define NK_HEAD (NF / 32)  /* 16 */

union HeadSM {
    struct { _Float16 A[2][128 * LDSA]; _Float16 B[2][64 * LDSA]; } g;
    float tile[128 * NHEAD];     /* 32 KB, reused after GEMM */
};

__global__ __launch_bounds__(256) void head_gemm_kernel(const _Float16* __restrict__ h16,
                                                        const _Float16* __restrict__ hwT,
                                                        const float* __restrict__ clsb,
                                                        const float* __restrict__ bboxb,
                                                        float* __restrict__ obj,
                                                        float* __restrict__ xx,
                                                        float* __restrict__ yy,
                                                        float* __restrict__ wd,
                                                        float* __restrict__ hd) {
    __shared__ HeadSM sm;

    const int tid   = threadIdx.x;
    const int mBase = blockIdx.x * 128;

    const int rA   = tid >> 1;
    const int offA = (tid & 1) * 16;
    const _Float16* aBase = h16 + (size_t)(mBase + rA) * NF + offA;

    const int nB   = tid >> 2;
    const int offB = (tid & 3) * 8;
    const _Float16* bBase = hwT + (size_t)nB * NF + offB;

    const int wave     = tid >> 5;
    const int lane     = tid & 31;
    const int mSubBase = (wave & 3) * 32;
    const int nSubBase = (wave >> 2) * 32;
    const int laneLo   = lane & 15;
    const int khalfA   = (lane >> 4) * 16;
    const int khalfB   = (lane >> 4) * 16;

    v8f c00 = {}, c01 = {}, c10 = {}, c11 = {};
    uint4 a0, a1, b0;

    /* stage 0 */
    a0 = *(const uint4*)aBase;
    a1 = *(const uint4*)(aBase + 8);
    b0 = *(const uint4*)bBase;
    store_stage(sm.g.A[0], sm.g.B[0], rA, offA, nB, offB, a0, a1, b0);
    __syncthreads();

    for (int ks = 0; ks < NK_HEAD; ++ks) {
        const int cur = ks & 1;
        const bool more = (ks + 1) < NK_HEAD;
        if (more) {
            const _Float16* ap = aBase + ((ks + 1) << 5);
            a0 = *(const uint4*)ap;
            a1 = *(const uint4*)(ap + 8);
            b0 = *(const uint4*)(bBase + ((ks + 1) << 5));
        }
        compute4(sm.g.A[cur], sm.g.B[cur], mSubBase, nSubBase, laneLo, khalfA, khalfB, c00, c01, c10, c11);
        if (more) store_stage(sm.g.A[cur ^ 1], sm.g.B[cur ^ 1], rA, offA, nB, offB, a0, a1, b0);
        __syncthreads();
    }

    /* write result tile (with bias) into LDS; GEMM buffers are dead now (post-barrier) */
    const int mHalf = (lane >> 4) * 8;
#pragma unroll
    for (int mi = 0; mi < 2; ++mi) {
#pragma unroll
        for (int ni = 0; ni < 2; ++ni) {
            const v8f acc = mi == 0 ? (ni == 0 ? c00 : c01) : (ni == 0 ? c10 : c11);
            const int nOut = nSubBase + ni * 16 + laneLo;   /* 0..63 */
            float bn = 0.f;
            if (nOut < 2 * KANCH)      bn = clsb[nOut];
            else if (nOut < 6 * KANCH) bn = bboxb[nOut - 2 * KANCH];
            const int r0 = mSubBase + mi * 16 + mHalf;
#pragma unroll
            for (int v = 0; v < 8; ++v)
                sm.tile[(r0 + v) * NHEAD + nOut] = acc[v] + bn;
        }
    }
    __syncthreads();

    /* fused decode: 128 pixels x 9 anchors */
    const float WS0[3] = {2.f, 4.f, 8.f};
    for (int i = tid; i < 128 * KANCH; i += 256) {
        const int row = i / KANCH;
        const int k   = i % KANCH;
        const float* trow = sm.tile + row * NHEAD;
        const float l0 = trow[k],             l1 = trow[KANCH + k];
        const float tx = trow[2 * KANCH + k], ty = trow[3 * KANCH + k];
        const float tw = trow[4 * KANCH + k], th = trow[5 * KANCH + k];

        const int m   = mBase + row;
        const int b   = m / (HFD * WFD);
        const int pix = m % (HFD * WFD);
        const int y   = pix / WFD;
        const int x   = pix % WFD;

        const float wk = WS0[k % 3], hk = WS0[k / 3];
        const float fx = (float)x, fy = (float)y;
        const bool valid = (fx - wk * 0.5f >= 0.f) && (fx + wk * 0.5f <= (float)WFD) &&
                           (fy - hk * 0.5f >= 0.f) && (fy + hk * 0.5f <= (float)HFD);

        const float om = 1.f / (1.f + expf(l0 - l1));   /* softmax([l0,l1])[1] */
        const size_t id = (size_t)b * NPROP + (size_t)pix * KANCH + k;
        obj[id] = valid ? om : 0.f;
        xx[id]  = fx - tx * wk;
        yy[id]  = fy - ty * hk;
        wd[id]  = wk * expf(fminf(fmaxf(tw, -20.f), 20.f));
        hd[id]  = hk * expf(fminf(fmaxf(th, -20.f), 20.f));
    }
}

/* ---------------- per-batch top-512 (threshold search + bitonic sort) ---------------- */

__global__ __launch_bounds__(1024) void topk_kernel(const float* __restrict__ obj,
                                                    const float* __restrict__ xx,
                                                    const float* __restrict__ yy,
                                                    const float* __restrict__ wd,
                                                    const float* __restrict__ hd,
                                                    float* __restrict__ out) {
    const int b = blockIdx.x;
    const int tid = threadIdx.x;
    const float* o = obj + (size_t)b * NPROP;

    __shared__ int cnt;
    __shared__ unsigned lo_s, hi_s;
    __shared__ int nHi, cEq;
    __shared__ float sv[NROI];
    __shared__ int   si[NROI];

    if (tid == 0) { lo_s = 0u; hi_s = 0x7F800000u; }
    __syncthreads();

    for (int it = 0; it < 31; ++it) {
        if (tid == 0) cnt = 0;
        __syncthreads();
        unsigned lo = lo_s, hi = hi_s;
        unsigned mid = lo + ((hi - lo) >> 1);
        int local = 0;
        for (int i = tid; i < NPROP; i += 1024)
            if (__float_as_uint(o[i]) > mid) local++;
        atomicAdd(&cnt, local);
        __syncthreads();
        if (tid == 0) {
            if (cnt >= NROI) lo_s = mid; else hi_s = mid;
        }
        __syncthreads();
    }
    const unsigned T = hi_s;

    if (tid < NROI) { sv[tid] = -1e30f; si[tid] = 0; }
    if (tid == 0) { nHi = 0; cEq = 0; }
    __syncthreads();

    for (int i = tid; i < NPROP; i += 1024) {
        if (__float_as_uint(o[i]) > T) {
            int pos = atomicAdd(&nHi, 1);
            if (pos < NROI) { sv[pos] = o[i]; si[pos] = i; }
        }
    }
    __syncthreads();
    int base = nHi; if (base > NROI) base = NROI;
    for (int i = tid; i < NPROP; i += 1024) {
        if (__float_as_uint(o[i]) == T) {
            int pos = base + atomicAdd(&cEq, 1);
            if (pos < NROI) { sv[pos] = o[i]; si[pos] = i; }
        }
    }
    __syncthreads();

    for (int ksz = 2; ksz <= NROI; ksz <<= 1) {
        for (int j = ksz >> 1; j > 0; j >>= 1) {
            if (tid < NROI) {
                int ixj = tid ^ j;
                if (ixj > tid) {
                    bool desc = ((tid & ksz) == 0);
                    float a = sv[tid], c = sv[ixj];
                    bool doswap = desc ? (a < c) : (a > c);
                    if (doswap) {
                        sv[tid] = c; sv[ixj] = a;
                        int t = si[tid]; si[tid] = si[ixj]; si[ixj] = t;
                    }
                }
            }
            __syncthreads();
        }
    }

    if (tid < NROI) {
        int idx = si[tid];
        size_t ob = ((size_t)b * NROI + tid) * 4;
        size_t ib = (size_t)b * NPROP + idx;
        out[ob + 0] = xx[ib];
        out[ob + 1] = yy[ib];
        out[ob + 2] = wd[ib];
        out[ob + 3] = hd[ib];
    }
}

/* ---------------- launch ---------------- */

extern "C" void kernel_launch(void* const* d_in, const int* in_sizes, int n_in,
                              void* d_out, int out_size, void* d_ws, size_t ws_size,
                              hipStream_t stream) {
    const float* features = (const float*)d_in[0];
    const float* conv1_w  = (const float*)d_in[1];
    const float* conv1_b  = (const float*)d_in[2];
    const float* cls_w    = (const float*)d_in[3];
    const float* cls_b    = (const float*)d_in[4];
    const float* bbox_w   = (const float*)d_in[5];
    const float* bbox_b   = (const float*)d_in[6];
    float* out = (float*)d_out;

    char* ws = (char*)d_ws;
    size_t off = 0;
    auto take = [&](size_t bytes) -> void* {
        void* p = ws + off;
        off = (off + bytes + 255) & ~(size_t)255;
        return p;
    };

    const size_t nPad = (size_t)BATCHN * HP * WP * CIN;   /* 33,587,200 */
    _Float16* featP = (_Float16*)take(nPad * 2);
    _Float16* wT16  = (_Float16*)take((size_t)NF * K_CONV * 2);
    _Float16* hwT16 = (_Float16*)take((size_t)NHEAD * NF * 2);
    _Float16* h16   = (_Float16*)take((size_t)M_TOTAL * NF * 2);
    float*    obj   = (float*)take((size_t)BATCHN * NPROP * 4);
    float*    xx    = (float*)take((size_t)BATCHN * NPROP * 4);
    float*    yy    = (float*)take((size_t)BATCHN * NPROP * 4);
    float*    wd    = (float*)take((size_t)BATCHN * NPROP * 4);
    float*    hd    = (float*)take((size_t)BATCHN * NPROP * 4);

    pad_feat_kernel<<<(int)((nPad + 255) / 256), 256, 0, stream>>>(features, featP, (int)nPad);
    pack_convw_kernel<<<(9 * CIN * NF + 255) / 256, 256, 0, stream>>>(conv1_w, wT16, 9 * CIN * NF);
    pack_headw_kernel<<<(NF * NHEAD + 255) / 256, 256, 0, stream>>>(cls_w, bbox_w, hwT16);

    conv_gemm_kernel<<<dim3(M_TOTAL / 128, NF / 64), 256, 0, stream>>>(featP, wT16, conv1_b, h16);
    head_gemm_kernel<<<dim3(M_TOTAL / 128), 256, 0, stream>>>(h16, hwT16, cls_b, bbox_b,
                                                              obj, xx, yy, wd, hd);
    topk_kernel<<<BATCHN, 1024, 0, stream>>>(obj, xx, yy, wd, hd, out);
}
